// SampledSofmax_33414845563312
// MI455X (gfx1250) — matrix operand order repaired and checked by reading, not compiled, not run on hardware
//
#include <hip/hip_runtime.h>

// Problem constants (match reference)
#define UNITS 128000
#define NEG   8192
#define NN    8192
#define CC    1024

typedef __attribute__((ext_vector_type(16))) __bf16 v16bf;
typedef __attribute__((ext_vector_type(8)))  float  v8f;
typedef unsigned short u16;

// -log(expected_count) for the log-uniform (Zipfian) candidate sampler.
__device__ __forceinline__ float neg_log_expected(int id) {
  float idf = (float)id;
  float p = (logf(idf + 2.0f) - logf(idf + 1.0f)) * (1.0f / logf((float)UNITS + 1.0f));
  float ec = -expm1f((float)NEG * log1pf(-p));
  return -logf(ec);
}

// Split f32 into bf16 hi (RNE) + bf16 lo (RNE of residual). a ~= hi + lo;
// 3 bf16 WMMA products (ah*bh + ah*bl + al*bh) recover ~fp32 GEMM accuracy.
__device__ __forceinline__ void split_bf16(float f, u16& h, u16& l) {
  unsigned u  = __float_as_uint(f);
  unsigned hr = (u + 0x7FFFu + ((u >> 16) & 1u)) >> 16;
  h = (u16)hr;
  float lo = f - __uint_as_float(hr << 16);
  unsigned ul = __float_as_uint(lo);
  l = (u16)((ul + 0x7FFFu + ((ul >> 16) & 1u)) >> 16);
}

// Convert 8 consecutive f32 -> packed hi/lo bf16 (16B each).
__device__ __forceinline__ void conv8(const float* __restrict__ src,
                                      u16* __restrict__ hi, u16* __restrict__ lo) {
  float4 f0 = *(const float4*)src;
  float4 f1 = *(const float4*)(src + 4);
  float f[8] = {f0.x, f0.y, f0.z, f0.w, f1.x, f1.y, f1.z, f1.w};
  unsigned hp[4], lp[4];
#pragma unroll
  for (int j = 0; j < 4; ++j) {
    u16 h0, l0, h1, l1;
    split_bf16(f[2 * j], h0, l0);
    split_bf16(f[2 * j + 1], h1, l1);
    hp[j] = (unsigned)h0 | ((unsigned)h1 << 16);
    lp[j] = (unsigned)l0 | ((unsigned)l1 << 16);
  }
  *(uint4*)hi = make_uint4(hp[0], hp[1], hp[2], hp[3]);
  *(uint4*)lo = make_uint4(lp[0], lp[1], lp[2], lp[3]);
}

// CDNA5 async global->LDS copy (ASYNCcnt-tracked). lds_off = LDS byte address
// (low 32 bits of the generic pointer per the flat-aperture rule).
__device__ __forceinline__ void async_b128(unsigned lds_off, const void* g) {
  asm volatile("global_load_async_to_lds_b128 %0, %1, off"
               :: "v"(lds_off), "v"((unsigned long long)(uintptr_t)g)
               : "memory");
}
__device__ __forceinline__ void wait_async0() {
  asm volatile("s_wait_asynccnt 0x0" ::: "memory");
}

// ---------------- K0: corrected sampled bias ----------------
__global__ void prep_cb(const int* __restrict__ sampled, const float* __restrict__ bias,
                        float* __restrict__ cb) {
  int i = blockIdx.x * blockDim.x + threadIdx.x;
  if (i < NEG) {
    int id = sampled[i];
    cb[i] = bias[id] + neg_log_expected(id);
  }
}

// ---------------- K1: true logits (f32 row dot) ----------------
__global__ void true_logits_k(const float* __restrict__ inputs, const int* __restrict__ targets,
                              const float* __restrict__ kernelW, const float* __restrict__ bias,
                              float* __restrict__ tl) {
  __shared__ float red[256];
  int n = blockIdx.x;
  int t = threadIdx.x;
  int tgt = targets[n];
  const float* a = inputs  + (size_t)n   * CC;
  const float* w = kernelW + (size_t)tgt * CC;
  float s = 0.f;
  for (int j = t; j < CC; j += 256) s += a[j] * w[j];
  red[t] = s;
  __syncthreads();
  for (int off = 128; off > 0; off >>= 1) {
    if (t < off) red[t] += red[t + off];
    __syncthreads();
  }
  if (t == 0) tl[n] = red[0] + bias[tgt] + neg_log_expected(tgt);
}

// ---------------- pre-conversion kernels (fast path) ----------------
__global__ void conv_a_k(const float* __restrict__ inputs,
                         u16* __restrict__ Ah, u16* __restrict__ Al) {
  size_t i = ((size_t)blockIdx.x * 256 + threadIdx.x) * 8;  // N*C/8 threads
  conv8(inputs + i, Ah + i, Al + i);
}
__global__ void conv_b_k(const float* __restrict__ kernelW, const int* __restrict__ sampled,
                         u16* __restrict__ Bh, u16* __restrict__ Bl) {
  unsigned gid = blockIdx.x * 256 + threadIdx.x;            // NEG*C/8 threads
  int s = gid >> 7;                                          // 128 threads per row
  int col = (gid & 127) * 8;
  const float* src = kernelW + (size_t)sampled[s] * CC + col;
  size_t dst = (size_t)s * CC + col;
  conv8(src, Bh + dst, Bl + dst);
}

// ---------------- WMMA fragment loads ----------------
#define LDK 40  // bf16 elements per LDS row (32 + 8 pad, keeps 16B alignment)

// A-matrix 16x32 bf16 frag (ISA 7.12.2): lane holds row M=lane&15; two 16B
// chunks at K = 8*half and K = 16 + 8*half.
__device__ __forceinline__ v16bf load_a_frag(const u16* base, int row0, int lane) {
  int half = lane >> 4, r = lane & 15;
  const u16* p = base + (row0 + r) * LDK + 8 * half;
  struct P { uint4 a, b; } t;
  t.a = *(const uint4*)(p);
  t.b = *(const uint4*)(p + 16);
  return __builtin_bit_cast(v16bf, t);
}
// B-matrix 32x16 bf16 frag: lane holds column N=lane&15 (== row of W since
// B = W^T); two 16B chunks at K = 8*half and K = 16 + 8*half.
__device__ __forceinline__ v16bf load_b_frag(const u16* base, int col0, int lane) {
  int half = lane >> 4, r = lane & 15;
  const u16* p = base + (col0 + r) * LDK + 8 * half;
  struct P { uint4 a, b; } t;
  t.a = *(const uint4*)(p);
  t.b = *(const uint4*)(p + 16);
  return __builtin_bit_cast(v16bf, t);
}

// Shared epilogue: adjust logits, mask accidental hits, per-row (max, sumexp).
__device__ __forceinline__ void lse_epilogue(float* Lg, v8f acc[2][2],
                                             const int* tgt_sh, const int* smp_sh,
                                             const float* cb_sh, int tid, int lane,
                                             int wm, int wn, int rowBase, int sChunk,
                                             float* pmax, float* psum) {
  int half = lane >> 4, nl = lane & 15;
#pragma unroll
  for (int tm = 0; tm < 2; ++tm)
#pragma unroll
    for (int tn = 0; tn < 2; ++tn) {
      int col = wn * 32 + tn * 16 + nl;
      float cbv = cb_sh[col];
      int   sid = smp_sh[col];
#pragma unroll
      for (int v = 0; v < 8; ++v) {
        int row = wm * 32 + tm * 16 + v + 8 * half;  // C/D layout: M = v + 8*half
        float val = acc[tm][tn][v] + cbv;
        if (sid == tgt_sh[row]) val = -3.0e38f;      // accidental-hit mask
        Lg[row * 65 + col] = val;
      }
    }
  __syncthreads();
  if (tid < 128) {
    const float* rowp = Lg + tid * 65;
    float m = -3.4e38f;
    for (int c = 0; c < 64; ++c) m = fmaxf(m, rowp[c]);
    float s = 0.f;
    for (int c = 0; c < 64; ++c) s += expf(rowp[c] - m);  // masked -> exp underflows to 0
    int rowG = rowBase + tid;
    pmax[(size_t)sChunk * NN + rowG] = m;   // [chunk][row]: coalesced both ends
    psum[(size_t)sChunk * NN + rowG] = s;
  }
}

// ---------------- K2a (fast): pre-converted operands, async double-buffered ----------------
#define STAGE_BYTES 30720  // Ah 10240 | Al 10240 | Bh 5120 | Bl 5120 (LDK rows)

__global__ __launch_bounds__(256)
void gemm_lse_fast(const u16* __restrict__ Ah, const u16* __restrict__ Al,
                   const u16* __restrict__ Bh, const u16* __restrict__ Bl,
                   const int* __restrict__ targets, const int* __restrict__ sampled,
                   const float* __restrict__ cb, float* __restrict__ pmax,
                   float* __restrict__ psum) {
  __shared__ __align__(16) char smem[2 * STAGE_BYTES];  // 61440B; logits tile reuses it
  __shared__ int   tgt_sh[128];
  __shared__ int   smp_sh[64];
  __shared__ float cb_sh[64];

  const int tid  = threadIdx.x;
  const int lane = tid & 31;
  const int w    = tid >> 5;
  const int wm   = w & 3;
  const int wn   = w >> 2;
  const int rowBase = blockIdx.x * 128;
  const int sBase   = blockIdx.y * 64;

  if (tid < 128) tgt_sh[tid] = targets[rowBase + tid];
  if (tid < 64) { smp_sh[tid] = sampled[sBase + tid]; cb_sh[tid] = cb[sBase + tid]; }

  // Per-thread copy slot: row r = tid>>2 (0..63), 16B quad q = tid&3.
  const int r = tid >> 2, q = tid & 3;
  const unsigned ldsBase = (unsigned)(uintptr_t)(void*)smem;  // LDS byte address
  const unsigned offA0 = (unsigned)(r * 80 + q * 16);         // LDK row = 80B
  const unsigned offA1 = offA0 + 64u * 80u;
  const size_t gA0 = (size_t)(rowBase + r)      * CC + q * 8;
  const size_t gA1 = (size_t)(rowBase + 64 + r) * CC + q * 8;
  const size_t gB  = (size_t)(sBase + r)        * CC + q * 8;

  v8f acc[2][2] = {{{}, {}}, {{}, {}}};

  // Prologue: async-copy tile 0 into buffer 0.
  {
    unsigned sb = ldsBase;
    async_b128(sb +     0 + offA0, Ah + gA0);
    async_b128(sb +     0 + offA1, Ah + gA1);
    async_b128(sb + 10240 + offA0, Al + gA0);
    async_b128(sb + 10240 + offA1, Al + gA1);
    async_b128(sb + 20480 + offA0, Bh + gB);
    async_b128(sb + 25600 + offA0, Bl + gB);
  }

  int buf = 0;
  for (int k = 0; k < 32; ++k) {
    wait_async0();      // this wave's tile-k copies landed in LDS
    __syncthreads();    // everyone's landed; previous compute finished
    if (k + 1 < 32) {   // prefetch tile k+1 into the other buffer
      int kb = (k + 1) * 32;
      unsigned sb = ldsBase + (unsigned)((buf ^ 1) * STAGE_BYTES);
      async_b128(sb +     0 + offA0, Ah + gA0 + kb);
      async_b128(sb +     0 + offA1, Ah + gA1 + kb);
      async_b128(sb + 10240 + offA0, Al + gA0 + kb);
      async_b128(sb + 10240 + offA1, Al + gA1 + kb);
      async_b128(sb + 20480 + offA0, Bh + gB + kb);
      async_b128(sb + 25600 + offA0, Bl + gB + kb);
    }
    const u16* As_hi = (const u16*)(smem + buf * STAGE_BYTES);
    const u16* As_lo = (const u16*)(smem + buf * STAGE_BYTES + 10240);
    const u16* Bs_hi = (const u16*)(smem + buf * STAGE_BYTES + 20480);
    const u16* Bs_lo = (const u16*)(smem + buf * STAGE_BYTES + 25600);

    v16bf ah[2], al[2], bh[2], bl[2];
#pragma unroll
    for (int tm = 0; tm < 2; ++tm) {
      ah[tm] = load_a_frag(As_hi, wm * 32 + tm * 16, lane);
      al[tm] = load_a_frag(As_lo, wm * 32 + tm * 16, lane);
    }
#pragma unroll
    for (int tn = 0; tn < 2; ++tn) {
      bh[tn] = load_b_frag(Bs_hi, wn * 32 + tn * 16, lane);
      bl[tn] = load_b_frag(Bs_lo, wn * 32 + tn * 16, lane);
    }
#pragma unroll
    for (int tm = 0; tm < 2; ++tm)
#pragma unroll
      for (int tn = 0; tn < 2; ++tn) {
        acc[tm][tn] = __builtin_amdgcn_wmma_f32_16x16x32_bf16(
            false, ah[tm], false, bh[tn], (short)0, acc[tm][tn], false, false);
        acc[tm][tn] = __builtin_amdgcn_wmma_f32_16x16x32_bf16(
            false, ah[tm], false, bl[tn], (short)0, acc[tm][tn], false, false);
        acc[tm][tn] = __builtin_amdgcn_wmma_f32_16x16x32_bf16(
            false, al[tm], false, bh[tn], (short)0, acc[tm][tn], false, false);
      }
    buf ^= 1;
  }

  __syncthreads();  // staging LDS dead; reuse as 128x65 f32 logits tile
  lse_epilogue((float*)smem, acc, tgt_sh, smp_sh, cb_sh, tid, lane, wm, wn,
               rowBase, blockIdx.y, pmax, psum);
}

// ---------------- K2b (fallback): convert-on-the-fly, single-buffered ----------------
__global__ __launch_bounds__(256)
void gemm_lse_fly(const float* __restrict__ inputs, const int* __restrict__ targets,
                  const int* __restrict__ sampled, const float* __restrict__ kernelW,
                  const float* __restrict__ cb, float* __restrict__ pmax,
                  float* __restrict__ psum) {
  __shared__ __align__(16) char smem[33280];
  __shared__ int   tgt_sh[128];
  __shared__ int   smp_sh[64];
  __shared__ float cb_sh[64];

  u16* As_hi = (u16*)smem;
  u16* As_lo = As_hi + 128 * LDK;
  u16* Bs_hi = As_lo + 128 * LDK;
  u16* Bs_lo = Bs_hi + 64 * LDK;

  const int tid  = threadIdx.x;
  const int lane = tid & 31;
  const int w    = tid >> 5;
  const int wm   = w & 3;
  const int wn   = w >> 2;
  const int rowBase = blockIdx.x * 128;
  const int sBase   = blockIdx.y * 64;

  if (tid < 128) tgt_sh[tid] = targets[rowBase + tid];
  if (tid < 64) { smp_sh[tid] = sampled[sBase + tid]; cb_sh[tid] = cb[sBase + tid]; }

  v8f acc[2][2] = {{{}, {}}, {{}, {}}};

  for (int kb = 0; kb < CC; kb += 32) {
    __syncthreads();
    for (int i = tid; i < 128 * 32; i += 256) {
      int r = i >> 5, c = i & 31;
      float f = inputs[(size_t)(rowBase + r) * CC + kb + c];
      u16 h, l; split_bf16(f, h, l);
      As_hi[r * LDK + c] = h;  As_lo[r * LDK + c] = l;
    }
    for (int i = tid; i < 64 * 32; i += 256) {
      int r = i >> 5, c = i & 31;
      float f = kernelW[(size_t)smp_sh[r] * CC + kb + c];
      u16 h, l; split_bf16(f, h, l);
      Bs_hi[r * LDK + c] = h;  Bs_lo[r * LDK + c] = l;
    }
    __syncthreads();

    v16bf ah[2], al[2], bh[2], bl[2];
#pragma unroll
    for (int tm = 0; tm < 2; ++tm) {
      ah[tm] = load_a_frag(As_hi, wm * 32 + tm * 16, lane);
      al[tm] = load_a_frag(As_lo, wm * 32 + tm * 16, lane);
    }
#pragma unroll
    for (int tn = 0; tn < 2; ++tn) {
      bh[tn] = load_b_frag(Bs_hi, wn * 32 + tn * 16, lane);
      bl[tn] = load_b_frag(Bs_lo, wn * 32 + tn * 16, lane);
    }
#pragma unroll
    for (int tm = 0; tm < 2; ++tm)
#pragma unroll
      for (int tn = 0; tn < 2; ++tn) {
        acc[tm][tn] = __builtin_amdgcn_wmma_f32_16x16x32_bf16(
            false, ah[tm], false, bh[tn], (short)0, acc[tm][tn], false, false);
        acc[tm][tn] = __builtin_amdgcn_wmma_f32_16x16x32_bf16(
            false, ah[tm], false, bl[tn], (short)0, acc[tm][tn], false, false);
        acc[tm][tn] = __builtin_amdgcn_wmma_f32_16x16x32_bf16(
            false, al[tm], false, bh[tn], (short)0, acc[tm][tn], false, false);
      }
  }

  __syncthreads();
  lse_epilogue((float*)smem, acc, tgt_sh, smp_sh, cb_sh, tid, lane, wm, wn,
               rowBase, blockIdx.y, pmax, psum);
}

// ---------------- K3: merge chunk partials + true logit per row ----------------
__global__ void row_reduce_k(const float* __restrict__ pmax, const float* __restrict__ psum,
                             const float* __restrict__ tl, float* __restrict__ per_ex) {
  int row = blockIdx.x * blockDim.x + threadIdx.x;
  if (row >= NN) return;
  float t = tl[row];
  float M = t;
  for (int i = 0; i < 128; ++i) M = fmaxf(M, pmax[(size_t)i * NN + row]);
  float d = expf(t - M);
  for (int i = 0; i < 128; ++i)
    d += psum[(size_t)i * NN + row] * expf(pmax[(size_t)i * NN + row] - M);
  per_ex[row] = logf(d) + M - t;   // logsumexp(all) - true_logit
}

// ---------------- K4: deterministic mean ----------------
__global__ void mean_k(const float* __restrict__ per_ex, float* __restrict__ out) {
  __shared__ float red[256];
  int t = threadIdx.x;
  float s = 0.f;
  for (int i = t; i < NN; i += 256) s += per_ex[i];
  red[t] = s;
  __syncthreads();
  for (int off = 128; off > 0; off >>= 1) {
    if (t < off) red[t] += red[t + off];
    __syncthreads();
  }
  if (t == 0) out[0] = red[0] * (1.0f / (float)NN);
}

extern "C" void kernel_launch(void* const* d_in, const int* in_sizes, int n_in,
                              void* d_out, int out_size, void* d_ws, size_t ws_size,
                              hipStream_t stream) {
  const float* inputs  = (const float*)d_in[0];
  const int*   targets = (const int*)  d_in[1];
  const int*   sampled = (const int*)  d_in[2];
  const float* kernelW = (const float*)d_in[3];
  const float* bias    = (const float*)d_in[4];
  float* out = (float*)d_out;

  // Workspace layout: f32 partials (~8.5 MB) then optional bf16 operands (64 MB)
  float* cb     = (float*)d_ws;                 // NEG
  float* tl     = cb + NEG;                     // NN
  float* pmax   = tl + NN;                      // 128 * NN
  float* psum   = pmax + (size_t)128 * NN;      // 128 * NN
  float* per_ex = psum + (size_t)128 * NN;      // NN
  size_t base_bytes = ((size_t)NEG + NN + 2ull * 128 * NN + NN) * 4;  // 8,486,912
  size_t op_elems   = (size_t)NN * CC;                                 // 8M
  size_t need       = base_bytes + 4ull * op_elems * sizeof(u16);      // ~75.6 MB
  u16* Ah = (u16*)((char*)d_ws + base_bytes);
  u16* Al = Ah + op_elems;
  u16* Bh = Al + op_elems;
  u16* Bl = Bh + op_elems;

  prep_cb<<<(NEG + 255) / 256, 256, 0, stream>>>(sampled, bias, cb);
  true_logits_k<<<NN, 256, 0, stream>>>(inputs, targets, kernelW, bias, tl);

  dim3 g(64, 128);  // 64 row-tiles x 128 s-chunks = 8192 blocks
  if (ws_size >= need) {
    // Fast path: pre-convert f32 -> (hi,lo) bf16 once; GEMM stages via async copies.
    int cblocks = (int)(op_elems / 8 / 256);  // 4096
    conv_a_k<<<cblocks, 256, 0, stream>>>(inputs, Ah, Al);
    conv_b_k<<<cblocks, 256, 0, stream>>>(kernelW, sampled, Bh, Bl);
    gemm_lse_fast<<<g, 256, 0, stream>>>(Ah, Al, Bh, Bl, targets, sampled, cb, pmax, psum);
  } else {
    gemm_lse_fly<<<g, 256, 0, stream>>>(inputs, targets, sampled, kernelW, cb, pmax, psum);
  }

  row_reduce_k<<<NN / 256, 256, 0, stream>>>(pmax, psum, tl, per_ex);
  mean_k<<<1, 256, 0, stream>>>(per_ex, out);
}
